// GAT_11038065951176
// MI455X (gfx1250) — compile-verified, matrix operand order checked by reading, not run on hardware
//
#include <hip/hip_runtime.h>
#include <hip/hip_bf16.h>
#include <math.h>

typedef __attribute__((ext_vector_type(16))) _Float16 v16h;
typedef __attribute__((ext_vector_type(8)))  _Float16 v8h;
typedef __attribute__((ext_vector_type(8)))  float    v8f;

// ---------------- problem constants ----------------
enum : int { Gc = 40, NPG = 500, NN = 20000, Eg = 320000, Et1 = 340000,
             FIN = 128, F1 = 512, HHc = 8, CCc = 64,
             K1c = 250, N1c = 10000, Et2 = 330000, NCc = 8, NP2 = 16,
             K2c = 125, N2c = 5000 };
#define NEGF (-1e30f)

// ---------------- workspace layout ----------------
static constexpr size_t AL(size_t x) { return (x + 255) & ~size_t(255); }
constexpr size_t O_MU1  = 0;
constexpr size_t O_VAR1 = AL(O_MU1  + (size_t)FIN * 4);
constexpr size_t O_XH   = AL(O_VAR1 + (size_t)FIN * 4);
constexpr size_t O_BF1  = AL(O_XH   + (size_t)NN * FIN * 2);
constexpr size_t O_HF   = AL(O_BF1  + (size_t)FIN * F1 * 2);
constexpr size_t O_AS1  = AL(O_HF   + (size_t)NN * F1 * 4);
constexpr size_t O_AD1  = AL(O_AS1  + (size_t)NN * HHc * 4);
constexpr size_t O_M1   = AL(O_AD1  + (size_t)NN * HHc * 4);
constexpr size_t O_DEN1 = AL(O_M1   + (size_t)NN * HHc * 4);
constexpr size_t O_OUT1 = AL(O_DEN1 + (size_t)NN * HHc * 4);
constexpr size_t O_NRM  = AL(O_OUT1 + (size_t)NN * F1 * 4);
constexpr size_t O_SC1  = AL(O_NRM  + 2 * 4);
constexpr size_t O_GID1 = AL(O_SC1  + (size_t)NN * 4);
constexpr size_t O_VAL1 = AL(O_GID1 + (size_t)N1c * 4);
constexpr size_t O_POS1 = AL(O_VAL1 + (size_t)N1c * 4);
constexpr size_t O_XP1  = AL(O_POS1 + (size_t)NN * 4);
constexpr size_t O_MU2  = AL(O_XP1  + (size_t)N1c * F1 * 4);
constexpr size_t O_VAR2 = AL(O_MU2  + (size_t)F1 * 4);
constexpr size_t O_X2H  = AL(O_VAR2 + (size_t)F1 * 4);
constexpr size_t O_BF2  = AL(O_X2H  + (size_t)N1c * F1 * 2);
constexpr size_t O_H2   = AL(O_BF2  + (size_t)F1 * NP2 * 2);
constexpr size_t O_AS2  = AL(O_H2   + (size_t)N1c * NP2 * 4);
constexpr size_t O_AD2  = AL(O_AS2  + (size_t)N1c * 4);
constexpr size_t O_M2   = AL(O_AD2  + (size_t)N1c * 4);
constexpr size_t O_DEN2 = AL(O_M2   + (size_t)N1c * 4);
constexpr size_t O_W2E  = AL(O_DEN2 + (size_t)N1c * 4);
constexpr size_t O_OUT2 = AL(O_W2E  + (size_t)Et2 * 4);
constexpr size_t O_SC2  = AL(O_OUT2 + (size_t)N1c * NCc * 4);
constexpr size_t O_GID2 = AL(O_SC2  + (size_t)N1c * 4);
constexpr size_t O_VAL2 = AL(O_GID2 + (size_t)N2c * 4);
constexpr size_t O_POS2 = AL(O_VAL2 + (size_t)N2c * 4);
constexpr size_t O_XP2  = AL(O_POS2 + (size_t)N1c * 4);
// CSR for gather-based conv1 aggregation (no float atomics)
constexpr size_t O_DEG  = AL(O_XP2  + (size_t)N2c * NCc * 4);
constexpr size_t O_OFF  = AL(O_DEG  + (size_t)NN * 4);
constexpr size_t O_CUR  = AL(O_OFF  + (size_t)(NN + 1) * 4);
constexpr size_t O_ELST = AL(O_CUR  + (size_t)NN * 4);

// ---------------- helpers ----------------
__device__ __forceinline__ float blockReduceSum(float v, float* sh) {
  int t = threadIdx.x;
  sh[t] = v; __syncthreads();
  for (int s = blockDim.x >> 1; s > 0; s >>= 1) {
    if (t < s) sh[t] += sh[t + s];
    __syncthreads();
  }
  float r = sh[0]; __syncthreads();
  return r;
}

__device__ __forceinline__ void atomicMaxF(float* addr, float val) {
  if (val >= 0.0f) atomicMax((int*)addr, __float_as_int(val));
  else             atomicMin((unsigned int*)addr, __float_as_uint(val));
}

__device__ __forceinline__ void edge1(int e, const int* ei, int& s, int& d) {
  if (e < Eg) { s = ei[e]; d = ei[Eg + e]; }
  else        { s = d = e - Eg; }
}

__device__ __forceinline__ bool edge2(int e, const int* ei, const int* pos, int& s, int& d) {
  if (e < Eg) {
    int ps = pos[ei[e]], pd = pos[ei[Eg + e]];
    bool ok = (ps >= 0) && (pd >= 0);
    s = ok ? ps : 0; d = ok ? pd : 0;
    return ok;
  }
  s = d = e - Eg;   // self loop in pooled graph (e-Eg < N1c)
  return true;
}

// ---------------- generic kernels ----------------
__global__ void k_fill(float* p, float v, long n) {
  long i = (long)blockIdx.x * blockDim.x + threadIdx.x;
  if (i < n) p[i] = v;
}

__global__ void k_bn_stats(const float* __restrict__ x, float* mu, float* var,
                           int Nrows, int F) {
  __shared__ float sh[256];
  int f = blockIdx.x;
  float s = 0.f, s2 = 0.f;
  for (int r = threadIdx.x; r < Nrows; r += blockDim.x) {
    float v = x[(size_t)r * F + f];
    s += v; s2 += v * v;
  }
  float ts  = blockReduceSum(s,  sh);
  float ts2 = blockReduceSum(s2, sh);
  if (threadIdx.x == 0) {
    float m = ts / Nrows;
    mu[f] = m;
    var[f] = ts2 / Nrows - m * m;
  }
}

__global__ void k_bn_apply_f16(const float* __restrict__ x, const float* mu, const float* var,
                               const float* g, const float* b, _Float16* __restrict__ out,
                               long total, int F) {
  long i = (long)blockIdx.x * blockDim.x + threadIdx.x;
  if (i >= total) return;
  int f = (int)(i % F);
  float v = g[f] * (x[i] - mu[f]) * rsqrtf(var[f] + 1e-5f) + b[f];
  out[i] = (_Float16)v;
}

// Pack W [K, realN] into WMMA B-fragments: frag[((kt*Ntiles+nt)*32+lane)*16+j]
__global__ void k_pack_frag(const float* __restrict__ W, _Float16* __restrict__ frag,
                            int Ntiles, int realN, int total) {
  int tid = blockIdx.x * blockDim.x + threadIdx.x;
  if (tid >= total) return;
  int j = tid & 15;
  int l = (tid >> 4) & 31;
  int rest = tid >> 9;
  int nt = rest % Ntiles;
  int kt = rest / Ntiles;
  int k = kt * 32 + ((l >= 16) ? 8 : 0) + (j < 8 ? j : j + 8);
  int n = nt * 16 + (l & 15);
  float v = (n < realN) ? W[(size_t)k * realN + n] : 0.0f;
  frag[tid] = (_Float16)v;
}

// A [M,K] f16 row-major, B pre-packed fragments, C [M, Npad] f32
template <int NT>
__global__ __launch_bounds__(256)
void k_gemm_wmma(const _Float16* __restrict__ A, const _Float16* __restrict__ B,
                 float* __restrict__ C, int M, int K, int Ntiles, int Npad) {
  int gw   = (int)((blockIdx.x * blockDim.x + threadIdx.x) >> 5);
  int lane = threadIdx.x & 31;
  int ngroups = (Ntiles + NT - 1) / NT;
  int mtile = gw / ngroups;
  int ng    = gw % ngroups;
  if (mtile >= (M >> 4)) return;

  v8f acc[NT] = {};
  const _Float16* Ar = A + (size_t)(mtile * 16 + (lane & 15)) * K + ((lane >> 4) << 3);
  const _Float16* Bb = B + ((size_t)ng * NT * 32 + lane) * 16;
  const size_t bstride = (size_t)Ntiles * 32 * 16;
  int ktiles = K >> 5;

  for (int kt = 0; kt < ktiles; ++kt) {
    __builtin_prefetch(Ar + 32, 0, 1);     // global_prefetch_b8
    v8h a0 = *(const v8h*)(Ar);
    v8h a1 = *(const v8h*)(Ar + 16);
    Ar += 32;
    v16h a = __builtin_shufflevector(a0, a1, 0, 1, 2, 3, 4, 5, 6, 7,
                                             8, 9, 10, 11, 12, 13, 14, 15);
#pragma unroll
    for (int nt = 0; nt < NT; ++nt) {
      v16h bf = *(const v16h*)(Bb + (size_t)nt * 32 * 16);
      acc[nt] = __builtin_amdgcn_wmma_f32_16x16x32_f16(
          false, a, false, bf, (short)0, acc[nt], false, false);
    }
    Bb += bstride;
  }

  int rbase = mtile * 16 + ((lane >> 4) << 3);
  int cbase = ng * NT * 16 + (lane & 15);
#pragma unroll
  for (int nt = 0; nt < NT; ++nt) {
#pragma unroll
    for (int v = 0; v < 8; ++v)
      C[(size_t)(rbase + v) * Npad + cbase + nt * 16] = acc[nt][v];
  }
}

__global__ void k_att(const float* __restrict__ H, const float* __restrict__ asrc,
                      const float* __restrict__ adst, float* AS, float* AD,
                      int Nn, int Hh, int Cc, int rstride) {
  int i = blockIdx.x * blockDim.x + threadIdx.x;
  if (i >= Nn * Hh) return;
  int n = i / Hh, h = i % Hh;
  const float* hp = H + (size_t)n * rstride + h * Cc;
  float s = 0.f, d = 0.f;
  for (int c = 0; c < Cc; ++c) {
    float v = hp[c];
    s += v * asrc[h * Cc + c];
    d += v * adst[h * Cc + c];
  }
  AS[i] = s; AD[i] = d;
}

__global__ void k_mfix(float* p, int n) {
  int i = blockIdx.x * blockDim.x + threadIdx.x;
  if (i < n) p[i] = (p[i] <= NEGF * 0.5f) ? 0.0f : p[i];
}

// -------- conv1 edge kernels (H=8, mask all-true) --------
__global__ void k_edge_max1(const int* ei, const float* AS, const float* AD, float* M1) {
  int t = blockIdx.x * blockDim.x + threadIdx.x;
  if (t >= Et1 * 8) return;
  int e = t >> 3, h = t & 7, s, d;
  edge1(e, ei, s, d);
  float ev = AS[s * 8 + h] + AD[d * 8 + h];
  ev = ev >= 0.f ? ev : 0.2f * ev;
  atomicMaxF(&M1[d * 8 + h], ev);
}

__global__ void k_edge_w1(const int* ei, const float* AS, const float* AD,
                          const float* M1, float* DEN1, float* alpha) {
  int t = blockIdx.x * blockDim.x + threadIdx.x;
  if (t >= Et1 * 8) return;
  int e = t >> 3, h = t & 7, s, d;
  edge1(e, ei, s, d);
  float ev = AS[s * 8 + h] + AD[d * 8 + h];
  ev = ev >= 0.f ? ev : 0.2f * ev;
  float w = expf(ev - M1[d * 8 + h]);
  atomicAdd(&DEN1[d * 8 + h], w);
  alpha[t] = w;
}

__global__ void k_alpha1(const int* ei, const float* DEN1, float* alpha) {
  int t = blockIdx.x * blockDim.x + threadIdx.x;
  if (t >= Et1 * 8) return;
  int e = t >> 3, h = t & 7, s, d;
  edge1(e, ei, s, d);
  alpha[t] = alpha[t] / fmaxf(DEN1[d * 8 + h], 1e-16f);
}

// -------- CSR build (count / scan / fill) --------
__global__ void k_deg_count(const int* ei, int* deg) {
  int e = blockIdx.x * blockDim.x + threadIdx.x;
  if (e >= Et1) return;
  int s, d;
  edge1(e, ei, s, d);
  atomicAdd(&deg[d], 1);
}

__global__ void k_scan_off(const int* __restrict__ deg, int* __restrict__ off, int n) {
  __shared__ int sh[256];
  __shared__ int carry;
  if (threadIdx.x == 0) carry = 0;
  __syncthreads();
  for (int base = 0; base < n; base += 256) {
    int i = base + (int)threadIdx.x;
    int v = (i < n) ? deg[i] : 0;
    sh[threadIdx.x] = v;
    __syncthreads();
    for (int s2 = 1; s2 < 256; s2 <<= 1) {
      int t = (threadIdx.x >= (unsigned)s2) ? sh[threadIdx.x - s2] : 0;
      __syncthreads();
      sh[threadIdx.x] += t;
      __syncthreads();
    }
    int excl = sh[threadIdx.x] - v;
    if (i < n) off[i] = carry + excl;
    __syncthreads();
    if (threadIdx.x == 255) carry += sh[255];
    __syncthreads();
  }
  if (threadIdx.x == 0) off[n] = carry;
}

__global__ void k_csr_fill(const int* ei, const int* __restrict__ off,
                           int* cursor, int* elist) {
  int e = blockIdx.x * blockDim.x + threadIdx.x;
  if (e >= Et1) return;
  int s, d;
  edge1(e, ei, s, d);
  int p = atomicAdd(&cursor[d], 1);
  elist[off[d] + p] = e;
}

// Gather aggregation: block per dst node, registers accumulate, fused bias+relu.
__global__ __launch_bounds__(256)
void k_agg1_csr(const int* ei, const int* __restrict__ off, const int* __restrict__ elist,
                const float* __restrict__ alpha, const float* __restrict__ Hf,
                const float* __restrict__ b, float* __restrict__ OUT) {
  int dn = blockIdx.x;
  int c0 = threadIdx.x, c1 = threadIdx.x + 256;
  float a0 = 0.f, a1 = 0.f;
  int beg = off[dn], end = off[dn + 1];
  for (int idx = beg; idx < end; ++idx) {
    int e = elist[idx];
    int s, d;
    edge1(e, ei, s, d);
    const float* hs = Hf + (size_t)s * F1;
    a0 += alpha[e * 8 + (c0 >> 6)] * hs[c0];
    a1 += alpha[e * 8 + (c1 >> 6)] * hs[c1];
  }
  OUT[(size_t)dn * F1 + c0] = fmaxf(a0 + b[c0], 0.f);
  OUT[(size_t)dn * F1 + c1] = fmaxf(a1 + b[c1], 0.f);
}

// -------- pooling --------
__global__ void k_wnorm(const float* w, int len, float* out) {
  __shared__ float sh[256];
  float s = 0.f;
  for (int i = threadIdx.x; i < len; i += blockDim.x) { float v = w[i]; s += v * v; }
  float t = blockReduceSum(s, sh);
  if (threadIdx.x == 0) out[0] = sqrtf(t);
}

__global__ void k_score512(const float* __restrict__ X, const float* w,
                           const float* nrm, float* sc, int F) {
  __shared__ float sh[256];
  int n = blockIdx.x;
  float s = 0.f;
  for (int f = threadIdx.x; f < F; f += blockDim.x) s += X[(size_t)n * F + f] * w[f];
  float t = blockReduceSum(s, sh);
  if (threadIdx.x == 0) sc[n] = tanhf(t / nrm[0]);
}

__global__ void k_score8(const float* __restrict__ X, const float* w,
                         const float* nrm, float* sc, int Nn) {
  int n = blockIdx.x * blockDim.x + threadIdx.x;
  if (n >= Nn) return;
  float s = 0.f;
  for (int c = 0; c < 8; ++c) s += X[(size_t)n * 8 + c] * w[c];
  sc[n] = tanhf(s / nrm[0]);
}

__global__ void k_topk(const float* sc, int n, int k, int* gidx, float* vals, int* pos) {
  __shared__ float s[512];
  int g = blockIdx.x, t = threadIdx.x;
  if (t < n) s[t] = sc[g * n + t];
  __syncthreads();
  if (t < n) {
    float mine = s[t];
    int rank = 0;
    for (int j = 0; j < n; ++j) {
      float o = s[j];
      rank += (o > mine) || (o == mine && j < t);   // stable, matches lax.top_k ties
    }
    if (rank < k) {
      int ni = g * k + rank;
      gidx[ni] = g * n + t;
      vals[ni] = mine;
      pos[g * n + t] = ni;
    } else {
      pos[g * n + t] = -1;
    }
  }
}

__global__ void k_gather(const float* __restrict__ X, const int* gidx, const float* vals,
                         float* __restrict__ out, int F, long total) {
  long i = (long)blockIdx.x * blockDim.x + threadIdx.x;
  if (i >= total) return;
  int r = (int)(i / F), f = (int)(i % F);
  out[i] = X[(size_t)gidx[r] * F + f] * vals[r];
}

// -------- conv2 edge kernels (H=1, C=8, masked) --------
__global__ void k_edge_max2(const int* ei, const int* pos, const float* AS,
                            const float* AD, float* M2) {
  int e = blockIdx.x * blockDim.x + threadIdx.x;
  if (e >= Et2) return;
  int s, d;
  if (!edge2(e, ei, pos, s, d)) return;
  float ev = AS[s] + AD[d];
  ev = ev >= 0.f ? ev : 0.2f * ev;
  atomicMaxF(&M2[d], ev);
}

__global__ void k_edge_w2(const int* ei, const int* pos, const float* AS,
                          const float* AD, const float* M2, float* DEN2, float* W2e) {
  int e = blockIdx.x * blockDim.x + threadIdx.x;
  if (e >= Et2) return;
  int s, d;
  bool ok = edge2(e, ei, pos, s, d);
  float w = 0.f;
  if (ok) {
    float ev = AS[s] + AD[d];
    ev = ev >= 0.f ? ev : 0.2f * ev;
    w = expf(ev - M2[d]);
    atomicAdd(&DEN2[d], w);
  }
  W2e[e] = w;
}

__global__ void k_agg2(const int* ei, const int* pos, const float* __restrict__ W2e,
                       const float* __restrict__ DEN2, const float* __restrict__ H2,
                       float* __restrict__ OUT2) {
  int t = blockIdx.x * blockDim.x + threadIdx.x;
  if (t >= Et2 * 8) return;
  int e = t >> 3, c = t & 7, s, d;
  if (!edge2(e, ei, pos, s, d)) return;
  float alpha = W2e[e] / fmaxf(DEN2[d], 1e-16f);
  atomicAdd(&OUT2[(size_t)d * 8 + c], alpha * H2[(size_t)s * NP2 + c]);
}

__global__ void k_bias_gelu(float* p, const float* b, long total) {
  long i = (long)blockIdx.x * blockDim.x + threadIdx.x;
  if (i >= total) return;
  float v = p[i] + b[i & 7];
  p[i] = 0.5f * v * (1.0f + erff(v * 0.70710678118654752f));   // exact gelu
}

// -------- output kernels --------
__global__ void k_final(const float* __restrict__ xp2, float* out) {
  __shared__ float cls[8];
  int g = blockIdx.x, t = threadIdx.x;
  if (t < 8) {
    float s = 0.f;
    for (int i = 0; i < K2c; ++i) s += xp2[((size_t)g * K2c + i) * 8 + t];
    cls[t] = s / (float)K2c;
  }
  __syncthreads();
  if (t == 0) {
    float m = cls[0];
    for (int c = 1; c < 8; ++c) m = fmaxf(m, cls[c]);
    float se = 0.f;
    for (int c = 0; c < 8; ++c) se += expf(cls[c] - m);
    float lse = m + logf(se);
    for (int c = 0; c < 8; ++c) out[g * 8 + c] = cls[c] - lse;
  }
}

__global__ void k_write_ei(const int* ei, float* eo) {
  int e = blockIdx.x * blockDim.x + threadIdx.x;
  if (e >= Et1) return;
  int s, d;
  edge1(e, ei, s, d);
  eo[e] = (float)s;
  eo[Et1 + e] = (float)d;
}

// ---------------- host launch ----------------
extern "C" void kernel_launch(void* const* d_in, const int* in_sizes, int n_in,
                              void* d_out, int out_size, void* d_ws, size_t ws_size,
                              hipStream_t stream) {
  (void)in_sizes; (void)n_in; (void)out_size; (void)ws_size;

  const float* x    = (const float*)d_in[0];
  const int*   ei   = (const int*)d_in[1];
  const float* bn1g = (const float*)d_in[3];
  const float* bn1b = (const float*)d_in[4];
  const float* W1   = (const float*)d_in[5];
  const float* as1  = (const float*)d_in[6];
  const float* ad1  = (const float*)d_in[7];
  const float* b1   = (const float*)d_in[8];
  const float* p1w  = (const float*)d_in[9];
  const float* bn2g = (const float*)d_in[10];
  const float* bn2b = (const float*)d_in[11];
  const float* W2   = (const float*)d_in[12];
  const float* as2  = (const float*)d_in[13];
  const float* ad2  = (const float*)d_in[14];
  const float* b2   = (const float*)d_in[15];
  const float* p2w  = (const float*)d_in[16];

  char* ws = (char*)d_ws;
  float*    mu1  = (float*)(ws + O_MU1);
  float*    var1 = (float*)(ws + O_VAR1);
  _Float16* xh   = (_Float16*)(ws + O_XH);
  _Float16* bf1  = (_Float16*)(ws + O_BF1);
  float*    hf   = (float*)(ws + O_HF);
  float*    AS1  = (float*)(ws + O_AS1);
  float*    AD1  = (float*)(ws + O_AD1);
  float*    M1   = (float*)(ws + O_M1);
  float*    DEN1 = (float*)(ws + O_DEN1);
  float*    OUT1 = (float*)(ws + O_OUT1);
  float*    nrm  = (float*)(ws + O_NRM);
  float*    sc1  = (float*)(ws + O_SC1);
  int*      gid1 = (int*)(ws + O_GID1);
  float*    val1 = (float*)(ws + O_VAL1);
  int*      pos1 = (int*)(ws + O_POS1);
  float*    xp1  = (float*)(ws + O_XP1);
  float*    mu2  = (float*)(ws + O_MU2);
  float*    var2 = (float*)(ws + O_VAR2);
  _Float16* x2h  = (_Float16*)(ws + O_X2H);
  _Float16* bf2  = (_Float16*)(ws + O_BF2);
  float*    h2   = (float*)(ws + O_H2);
  float*    AS2  = (float*)(ws + O_AS2);
  float*    AD2  = (float*)(ws + O_AD2);
  float*    M2   = (float*)(ws + O_M2);
  float*    DEN2 = (float*)(ws + O_DEN2);
  float*    W2e  = (float*)(ws + O_W2E);
  float*    OUT2 = (float*)(ws + O_OUT2);
  float*    sc2  = (float*)(ws + O_SC2);
  int*      gid2 = (int*)(ws + O_GID2);
  float*    val2 = (float*)(ws + O_VAL2);
  int*      pos2 = (int*)(ws + O_POS2);
  float*    xp2  = (float*)(ws + O_XP2);
  int*      deg  = (int*)(ws + O_DEG);
  int*      off  = (int*)(ws + O_OFF);
  int*      cur  = (int*)(ws + O_CUR);
  int*      elst = (int*)(ws + O_ELST);

  float* out    = (float*)d_out;               // [0,320): logits
  float* ei_out = out + Gc * NCc;              // [320, 320+2*Et1): ei_aug as float
  float* alpha1 = ei_out + 2 * Et1;            // alpha1 [Et1, 8]

  auto nb = [](long long t, int b) { return (unsigned)((t + b - 1) / b); };

  // ---- BN1 + f16 convert ----
  k_bn_stats<<<FIN, 256, 0, stream>>>(x, mu1, var1, NN, FIN);
  k_bn_apply_f16<<<nb((long long)NN * FIN, 256), 256, 0, stream>>>(
      x, mu1, var1, bn1g, bn1b, xh, (long)NN * FIN, FIN);

  // ---- GEMM1: [20000,128] x [128,512] via WMMA ----
  k_pack_frag<<<nb(FIN * F1, 256), 256, 0, stream>>>(W1, bf1, F1 / 16, F1, FIN * F1);
  {
    int waves = (NN / 16) * ((F1 / 16) / 4);
    k_gemm_wmma<4><<<nb((long long)waves * 32, 256), 256, 0, stream>>>(
        xh, bf1, hf, NN, FIN, F1 / 16, F1);
  }

  // ---- CSR of augmented edges by destination (gather aggregation) ----
  k_fill<<<nb(NN, 256), 256, 0, stream>>>((float*)deg, 0.f, NN);
  k_deg_count<<<nb(Et1, 256), 256, 0, stream>>>(ei, deg);
  k_scan_off<<<1, 256, 0, stream>>>(deg, off, NN);
  k_fill<<<nb(NN, 256), 256, 0, stream>>>((float*)cur, 0.f, NN);
  k_csr_fill<<<nb(Et1, 256), 256, 0, stream>>>(ei, off, cur, elst);

  // ---- attention conv1 ----
  k_att<<<nb(NN * HHc, 256), 256, 0, stream>>>(hf, as1, ad1, AS1, AD1, NN, HHc, CCc, F1);
  k_fill<<<nb(NN * HHc, 256), 256, 0, stream>>>(M1, NEGF, (long)NN * HHc);
  k_fill<<<nb(NN * HHc, 256), 256, 0, stream>>>(DEN1, 0.f, (long)NN * HHc);
  k_edge_max1<<<nb((long long)Et1 * 8, 256), 256, 0, stream>>>(ei, AS1, AD1, M1);
  k_mfix<<<nb(NN * HHc, 256), 256, 0, stream>>>(M1, NN * HHc);
  k_edge_w1<<<nb((long long)Et1 * 8, 256), 256, 0, stream>>>(ei, AS1, AD1, M1, DEN1, alpha1);
  k_alpha1<<<nb((long long)Et1 * 8, 256), 256, 0, stream>>>(ei, DEN1, alpha1);
  // gather aggregation + fused bias + relu (no float atomics, no OUT1 prefill)
  k_agg1_csr<<<NN, 256, 0, stream>>>(ei, off, elst, alpha1, hf, b1, OUT1);

  // ---- pool1 ----
  k_wnorm<<<1, 256, 0, stream>>>(p1w, F1, nrm + 0);
  k_score512<<<NN, 256, 0, stream>>>(OUT1, p1w, nrm + 0, sc1, F1);
  k_topk<<<Gc, 512, 0, stream>>>(sc1, NPG, K1c, gid1, val1, pos1);
  k_gather<<<nb((long long)N1c * F1, 256), 256, 0, stream>>>(
      OUT1, gid1, val1, xp1, F1, (long)N1c * F1);

  // ---- BN2 + f16 convert ----
  k_bn_stats<<<F1, 256, 0, stream>>>(xp1, mu2, var2, N1c, F1);
  k_bn_apply_f16<<<nb((long long)N1c * F1, 256), 256, 0, stream>>>(
      xp1, mu2, var2, bn2g, bn2b, x2h, (long)N1c * F1, F1);

  // ---- GEMM2: [10000,512] x [512,8->16pad] via WMMA ----
  k_pack_frag<<<nb(F1 * NP2, 256), 256, 0, stream>>>(W2, bf2, 1, NCc, F1 * NP2);
  {
    int waves = (N1c / 16) * 1;
    k_gemm_wmma<1><<<nb((long long)waves * 32, 256), 256, 0, stream>>>(
        x2h, bf2, h2, N1c, F1, 1, NP2);
  }

  // ---- attention conv2 ----
  k_att<<<nb(N1c, 256), 256, 0, stream>>>(h2, as2, ad2, AS2, AD2, N1c, 1, NCc, NP2);
  k_fill<<<nb(N1c, 256), 256, 0, stream>>>(M2, NEGF, N1c);
  k_fill<<<nb(N1c, 256), 256, 0, stream>>>(DEN2, 0.f, N1c);
  k_fill<<<nb(N1c * NCc, 256), 256, 0, stream>>>(OUT2, 0.f, (long)N1c * NCc);
  k_edge_max2<<<nb(Et2, 256), 256, 0, stream>>>(ei, pos1, AS2, AD2, M2);
  k_mfix<<<nb(N1c, 256), 256, 0, stream>>>(M2, N1c);
  k_edge_w2<<<nb(Et2, 256), 256, 0, stream>>>(ei, pos1, AS2, AD2, M2, DEN2, W2e);
  k_agg2<<<nb((long long)Et2 * 8, 256), 256, 0, stream>>>(ei, pos1, W2e, DEN2, h2, OUT2);
  k_bias_gelu<<<nb(N1c * NCc, 256), 256, 0, stream>>>(OUT2, b2, (long)N1c * NCc);

  // ---- pool2 ----
  k_wnorm<<<1, 256, 0, stream>>>(p2w, NCc, nrm + 1);
  k_score8<<<nb(N1c, 256), 256, 0, stream>>>(OUT2, p2w, nrm + 1, sc2, N1c);
  k_topk<<<Gc, 512, 0, stream>>>(sc2, K1c, K2c, gid2, val2, pos2);
  k_gather<<<nb(N2c * NCc, 256), 256, 0, stream>>>(
      OUT2, gid2, val2, xp2, NCc, (long)N2c * NCc);

  // ---- mean pool + log_softmax + aux outputs ----
  k_final<<<Gc, 32, 0, stream>>>(xp2, out);
  k_write_ei<<<nb(Et1, 256), 256, 0, stream>>>(ei, ei_out);
}
